// MolMambaEncoder_24378234372313
// MI455X (gfx1250) — matrix operand order repaired and checked by reading, not compiled, not run on hardware
//
#include <hip/hip_runtime.h>
#include <hip/hip_bf16.h>

typedef __attribute__((ext_vector_type(16))) _Float16 v16h;
typedef __attribute__((ext_vector_type(8)))  _Float16 v8h;
typedef __attribute__((ext_vector_type(8)))  float    v8f;

#define NPG 128   // nodes per graph
#define EPG 2048  // edges per graph
#define NG  64    // graphs
#define NN  8192  // total nodes
#define NE  131072// total edges
#define HD  256   // hidden dim

__device__ __forceinline__ void pack8(v16h& f, int base, float4 u, float4 w) {
  f[base + 0] = (_Float16)u.x; f[base + 1] = (_Float16)u.y;
  f[base + 2] = (_Float16)u.z; f[base + 3] = (_Float16)u.w;
  f[base + 4] = (_Float16)w.x; f[base + 5] = (_Float16)w.y;
  f[base + 6] = (_Float16)w.z; f[base + 7] = (_Float16)w.w;
}

// ---------------------------------------------------------------------------
// WMMA GEMM: C[M,N] = act(A[M,K] @ Bt^T + bias).
//   A:  f32 [M,K] row-major (activations), converted in-register via cvt_pk.
//   Bt: f16 [N,K] row-major (pre-converted weights) -> pure b128 loads.
// Requires K % 32 == 0, N % 64 == 0, M % 16 == 0, 16B-aligned rows.
// One wave per 16x64 output strip: A fragment loaded once per K-step and
// reused across 4 accumulators (4 v_wmma per K-step).
// ---------------------------------------------------------------------------
template <bool RELU>
__global__ __launch_bounds__(32) void wmma_gemm_kernel(
    const float* __restrict__ A, const _Float16* __restrict__ Bt,
    const float* __restrict__ bias, float* __restrict__ C,
    int M, int N, int K)
{
  const int bm   = blockIdx.x * 16;
  const int bn   = blockIdx.y * 64;
  const int lane = threadIdx.x & 31;
  const int l16  = lane & 15;
  const int hi   = lane >> 4;   // 0 or 1
  const int kbA  = hi * 8;      // A fragment K base per lane half
  const int kbB  = hi * 8;      // B fragment K base per lane half
  const int mb   = hi * 8;      // C/D row base per lane half

  const float* Arow = A + (size_t)(bm + l16) * K;

  v8f acc[4] = {};
  for (int k0 = 0; k0 < K; k0 += 32) {
    // A fragment: lane holds row (bm+l16), K = {kbA..kbA+7, kbA+16..kbA+23}
    v16h a;
    {
      const float4* p0 = reinterpret_cast<const float4*>(Arow + k0 + kbA);
      const float4* p1 = reinterpret_cast<const float4*>(Arow + k0 + kbA + 16);
      pack8(a, 0, p0[0], p0[1]);
      pack8(a, 8, p1[0], p1[1]);
    }
#pragma unroll
    for (int j = 0; j < 4; ++j) {
      const int ncol = bn + j * 16 + l16;
      // lane holds column ncol: f16 K = {kbB..kbB+7, kbB+16..kbB+23} of row ncol
      const v8h* Bp =
          reinterpret_cast<const v8h*>(Bt + (size_t)ncol * K + k0 + kbB);
      v8h b0 = Bp[0], b1 = Bp[2];
      v16h b = __builtin_shufflevector(b0, b1, 0, 1, 2, 3, 4, 5, 6, 7,
                                       8, 9, 10, 11, 12, 13, 14, 15);
      acc[j] = __builtin_amdgcn_wmma_f32_16x16x32_f16(
          false, a, false, b, (short)0, acc[j], false, false);
    }
  }

#pragma unroll
  for (int j = 0; j < 4; ++j) {
    const int ncol = bn + j * 16 + l16;
    const float bv = bias ? bias[ncol] : 0.0f;
#pragma unroll
    for (int v = 0; v < 8; ++v) {
      float val = acc[j][v] + bv;
      if (RELU) val = fmaxf(val, 0.0f);
      C[(size_t)(bm + mb + v) * N + ncol] = val;
    }
  }
}

// Wt[n][k] = (f16) W[k][n], zero-padded for k in [Ksrc, Kdst). Grid = N.
__global__ void convert_transpose_kernel(const float* __restrict__ W,
                                         _Float16* __restrict__ Wt,
                                         int Ksrc, int Kdst, int N)
{
  int n = blockIdx.x;
  for (int k = threadIdx.x; k < Kdst; k += blockDim.x)
    Wt[(size_t)n * Kdst + k] =
        (k < Ksrc) ? (_Float16)W[(size_t)k * N + n] : (_Float16)0.0f;
}

// Straight f32 -> f16 conversion (weights already stored [N,K]).
__global__ void convert_f16_kernel(const float* __restrict__ W,
                                   _Float16* __restrict__ Wh, int total)
{
  int idx = blockIdx.x * 256 + threadIdx.x;
  if (idx < total) Wh[idx] = (_Float16)W[idx];
}

// Zero-pad edge_attr [NE,16] -> [NE,32] so the fast GEMM covers K=16.
__global__ void pad_edge_attr_kernel(const float* __restrict__ ea,
                                     float* __restrict__ pad)
{
  int e = blockIdx.x;
  int c = threadIdx.x;  // 32
  pad[(size_t)e * 32 + c] = (c < 16) ? ea[(size_t)e * 16 + c] : 0.0f;
}

// ---------------------------------------------------------------------------
// Per-graph BFS permutation (matches reference: start = argmax out-degree,
// adjacency in edge order, FIFO BFS, then unvisited in index order) and a
// deterministic CSR-by-destination for the ordered segment sum.
// ---------------------------------------------------------------------------
__global__ void bfs_csr_kernel(const int* __restrict__ ei, int* __restrict__ perm,
                               int* __restrict__ doff, int* __restrict__ dedge,
                               int* __restrict__ scratch)
{
  int g = blockIdx.x;
  if (threadIdx.x != 0) return;
  const int* src = ei;
  const int* dst = ei + NE;
  const int lo = g * NPG;
  const int eb = g * EPG;

  int* deg = scratch + g * 2688;
  int* off = deg + NPG;        // NPG+1
  int* cur = off + NPG + 1;    // NPG
  int* adj = cur + NPG;        // EPG
  int* vis = adj + EPG;        // NPG

  for (int i = 0; i < NPG; ++i) { deg[i] = 0; vis[i] = 0; }
  for (int e = 0; e < EPG; ++e) deg[src[eb + e] - lo]++;
  off[0] = 0;
  for (int i = 0; i < NPG; ++i) off[i + 1] = off[i] + deg[i];
  for (int i = 0; i < NPG; ++i) cur[i] = off[i];
  for (int e = 0; e < EPG; ++e) {
    int s = src[eb + e] - lo;
    adj[cur[s]++] = dst[eb + e] - lo;
  }
  int start = 0, best = deg[0];
  for (int i = 1; i < NPG; ++i) if (deg[i] > best) { best = deg[i]; start = i; }

  int* order = perm + lo;      // FIFO: dequeue order == enqueue order
  int head = 0, tail = 0;
  order[tail++] = start; vis[start] = 1;
  while (head < tail) {
    int u = order[head++];
    for (int j = off[u]; j < off[u + 1]; ++j) {
      int v = adj[j];
      if (!vis[v]) { vis[v] = 1; order[tail++] = v; }
    }
  }
  for (int i = 0; i < NPG; ++i) if (!vis[i]) order[tail++] = i;
  for (int i = 0; i < NPG; ++i) order[i] += lo;

  // CSR by destination, edge ids ascending (deterministic sum order)
  for (int i = 0; i < NPG; ++i) deg[i] = 0;
  for (int e = 0; e < EPG; ++e) deg[dst[eb + e] - lo]++;
  int* dof = doff + g * (NPG + 1);
  dof[0] = 0;
  for (int i = 0; i < NPG; ++i) dof[i + 1] = dof[i] + deg[i];
  for (int i = 0; i < NPG; ++i) cur[i] = dof[i];
  for (int e = 0; e < EPG; ++e) {
    int d = dst[eb + e] - lo;
    dedge[eb + cur[d]++] = eb + e;   // global edge id
  }
}

// ---------------------------------------------------------------------------
// z[v] = (1+eps)*h[v] + sum_{e: dst(e)=v} relu(h[src(e)] + edgefeat[e])
// One block per node, one thread per channel; ordered (deterministic) sum.
// Prefetch the next edge's feature cacheline — this kernel streams the
// 134MB edge buffer and is the bandwidth floor of the pipeline.
// ---------------------------------------------------------------------------
__global__ void gine_aggregate_kernel(const float* __restrict__ h,
                                      const float* __restrict__ e,
                                      const int* __restrict__ ei,
                                      const int* __restrict__ doff,
                                      const int* __restrict__ dedge,
                                      const float* __restrict__ eps,
                                      float* __restrict__ z)
{
  int vtx = blockIdx.x;
  int c = threadIdx.x;
  int g = vtx >> 7, loc = vtx & 127;
  const int* src = ei;
  int s0 = doff[g * (NPG + 1) + loc];
  int s1 = doff[g * (NPG + 1) + loc + 1];
  float acc = 0.0f;
  for (int j = s0; j < s1; ++j) {
    int eid = dedge[g * EPG + j];
    if (j + 1 < s1) {
      int eidn = dedge[g * EPG + j + 1];
      __builtin_prefetch(e + (size_t)eidn * HD + c, 0, 1);
    }
    int s = src[eid];
    acc += fmaxf(h[(size_t)s * HD + c] + e[(size_t)eid * HD + c], 0.0f);
  }
  z[(size_t)vtx * HD + c] = (1.0f + eps[0]) * h[(size_t)vtx * HD + c] + acc;
}

// BatchNorm stats per channel over all 8192 nodes (population variance).
__global__ void bn_stats_kernel(const float* __restrict__ z2, float* __restrict__ stat)
{
  int c = blockIdx.x;
  int tid = threadIdx.x;
  float s = 0.0f, sq = 0.0f;
  for (int r = tid; r < NN; r += 256) {
    float v = z2[(size_t)r * HD + c];
    s += v; sq += v * v;
  }
  __shared__ float ss[256], sqq[256];
  ss[tid] = s; sqq[tid] = sq;
  __syncthreads();
  for (int st = 128; st > 0; st >>= 1) {
    if (tid < st) { ss[tid] += ss[tid + st]; sqq[tid] += sqq[tid + st]; }
    __syncthreads();
  }
  if (tid == 0) {
    float mu = ss[0] * (1.0f / NN);
    float var = sqq[0] * (1.0f / NN) - mu * mu;
    stat[c] = mu;
    stat[HD + c] = rsqrtf(var + 1e-5f);
  }
}

// h = relu(gamma*(z2-mu)*rstd + beta) + h
__global__ void bn_res_kernel(const float* __restrict__ z2, const float* __restrict__ stat,
                              const float* __restrict__ gamma, const float* __restrict__ beta,
                              float* __restrict__ h)
{
  size_t idx = (size_t)blockIdx.x * 256 + threadIdx.x;
  int c = (int)(idx & (HD - 1));
  float v = gamma[c] * (z2[idx] - stat[c]) * stat[HD + c] + beta[c];
  h[idx] = fmaxf(v, 0.0f) + h[idx];
}

// X2[(t*64+g), :] = h[perm[g*128+t], :] + pe[t, :]   (time-major for the GRU)
__global__ void build_X_kernel(const float* __restrict__ h, const int* __restrict__ perm,
                               const float* __restrict__ pe, float* __restrict__ X2)
{
  int row = blockIdx.x;         // g*128 + i
  int c = threadIdx.x;
  int g = row >> 7, i = row & 127;
  int p = perm[row];
  X2[((size_t)(i * NG + g)) * HD + c] = h[(size_t)p * HD + c] + pe[(size_t)i * HD + c];
}

// One GRU step: gates from gx[t] (already has bih) and gh (already has bhh).
__global__ void gru_gate_kernel(const float* __restrict__ gxt, const float* __restrict__ gh,
                                float* __restrict__ h, float* __restrict__ hsum, int t)
{
  int m = blockIdx.x;           // graph
  int c = threadIdx.x;          // channel
  const float* gx = gxt + (size_t)m * 768;
  const float* G  = gh  + (size_t)m * 768;
  float r  = 1.0f / (1.0f + __expf(-(gx[c] + G[c])));
  float zg = 1.0f / (1.0f + __expf(-(gx[256 + c] + G[256 + c])));
  float nn = tanhf(gx[512 + c] + r * G[512 + c]);
  float hv = h[m * HD + c];
  float hn = (1.0f - zg) * nn + zg * hv;
  h[m * HD + c] = hn;
  hsum[m * HD + c] = (t == 0) ? hn : hsum[m * HD + c] + hn;
}

// fused = [last | mean_tok | graph-mean-pool(h)]
__global__ void assemble_fused_kernel(const float* __restrict__ h,
                                      const float* __restrict__ hstate,
                                      const float* __restrict__ hsum,
                                      float* __restrict__ fused)
{
  int g = blockIdx.x;
  int c = threadIdx.x;
  fused[(size_t)g * 768 + c] = hstate[g * HD + c];
  fused[(size_t)g * 768 + 256 + c] = hsum[g * HD + c] * (1.0f / NPG);
  float s = 0.0f;
  for (int i = 0; i < NPG; ++i) s += h[((size_t)g * NPG + i) * HD + c];
  fused[(size_t)g * 768 + 512 + c] = s * (1.0f / NPG);
}

// ---------------------------------------------------------------------------
extern "C" void kernel_launch(void* const* d_in, const int* in_sizes, int n_in,
                              void* d_out, int out_size, void* d_ws, size_t ws_size,
                              hipStream_t stream)
{
  (void)in_sizes; (void)n_in; (void)out_size; (void)ws_size;
  const float* x         = (const float*)d_in[0];
  const float* edge_attr = (const float*)d_in[1];
  const int*   edge_idx  = (const int*)d_in[2];
  const float* proj_in_W = (const float*)d_in[4];
  const float* proj_in_b = (const float*)d_in[5];
  const float* edge_W1   = (const float*)d_in[6];
  const float* edge_b1   = (const float*)d_in[7];
  const float* edge_W2   = (const float*)d_in[8];
  const float* edge_b2   = (const float*)d_in[9];
  const float* gine_W1   = (const float*)d_in[10];
  const float* gine_b1   = (const float*)d_in[11];
  const float* gine_W2   = (const float*)d_in[12];
  const float* gine_b2   = (const float*)d_in[13];
  const float* gine_eps  = (const float*)d_in[14];
  const float* bn_gamma  = (const float*)d_in[15];
  const float* bn_beta   = (const float*)d_in[16];
  const float* pe        = (const float*)d_in[17];
  const float* gru_Wih   = (const float*)d_in[18];
  const float* gru_Whh   = (const float*)d_in[19];
  const float* gru_bih   = (const float*)d_in[20];
  const float* gru_bhh   = (const float*)d_in[21];
  const float* out_W     = (const float*)d_in[22];
  const float* out_b     = (const float*)d_in[23];
  float* out = (float*)d_out;

  // ---- workspace layout: f32 region ----
  float* fws = (float*)d_ws;
  size_t off = 0;
  auto take = [&](size_t n) { float* p = fws + off; off += n; return p; };
  float* e_buf  = take((size_t)NE * HD);         // persistent edge features
  float* h_buf  = take((size_t)NN * HD);         // node features
  float* ea_pad = take((size_t)NE * 32);         // zero-padded edge_attr
  // union region: e1 (edge MLP hidden) dies before the per-layer buffers live
  size_t layer_elems = 4 * (size_t)NN * HD + (size_t)NN * 768;
  size_t region_elems = ((size_t)NE * HD > layer_elems) ? (size_t)NE * HD : layer_elems;
  float* region = take(region_elems);
  float* e1 = region;
  float* z  = region;
  float* t1 = z  + (size_t)NN * HD;
  float* z2 = t1 + (size_t)NN * HD;
  float* X2 = z2 + (size_t)NN * HD;
  float* gx = X2 + (size_t)NN * HD;              // [128][64][768] time-major
  float* stat   = take(2 * HD);
  float* hstate = take((size_t)NG * HD);
  float* gh     = take((size_t)NG * 768);
  float* hsum   = take((size_t)NG * HD);
  float* fused  = take((size_t)NG * 768);

  // ---- f16 weight region (16B aligned: all offsets multiples of 8 halves) --
  _Float16* hws = (_Float16*)(fws + off);
  size_t hoff = 0;
  auto takeh = [&](size_t n) { n = (n + 7) & ~(size_t)7;
                               _Float16* p = hws + hoff; hoff += n; return p; };
  _Float16* projWt  = takeh((size_t)HD * 64);    // [256][64]
  _Float16* edgeW1t = takeh((size_t)HD * 32);    // [256][32] (K padded 16->32)
  _Float16* edgeW2t = takeh((size_t)HD * HD);    // [256][256]
  _Float16* gineW1t = takeh((size_t)3 * HD * HD);
  _Float16* gineW2t = takeh((size_t)3 * HD * HD);
  _Float16* WihH    = takeh((size_t)768 * HD);   // already [N,K]
  _Float16* WhhH    = takeh((size_t)768 * HD);
  _Float16* outWt   = takeh((size_t)HD * 768);   // [256][768]

  // ---- int region ----
  int* iws     = (int*)(hws + hoff);
  int* perm    = iws;
  int* doff    = perm + NN;
  int* dedge   = doff + NG * (NPG + 1);
  int* scratch = dedge + NE;

  // 0) one-time weight conversion to f16 [N,K] + edge_attr K-padding
  convert_transpose_kernel<<<HD, 256, 0, stream>>>(proj_in_W, projWt, 64, 64, HD);
  convert_transpose_kernel<<<HD, 256, 0, stream>>>(edge_W1, edgeW1t, 16, 32, HD);
  convert_transpose_kernel<<<HD, 256, 0, stream>>>(edge_W2, edgeW2t, HD, HD, HD);
  for (int l = 0; l < 3; ++l) {
    convert_transpose_kernel<<<HD, 256, 0, stream>>>(
        gine_W1 + (size_t)l * HD * HD, gineW1t + (size_t)l * HD * HD, HD, HD, HD);
    convert_transpose_kernel<<<HD, 256, 0, stream>>>(
        gine_W2 + (size_t)l * HD * HD, gineW2t + (size_t)l * HD * HD, HD, HD, HD);
  }
  convert_f16_kernel<<<(768 * HD) / 256, 256, 0, stream>>>(gru_Wih, WihH, 768 * HD);
  convert_f16_kernel<<<(768 * HD) / 256, 256, 0, stream>>>(gru_Whh, WhhH, 768 * HD);
  convert_transpose_kernel<<<HD, 256, 0, stream>>>(out_W, outWt, 768, 768, HD);
  pad_edge_attr_kernel<<<NE, 32, 0, stream>>>(edge_attr, ea_pad);

  // 1) BFS permutation + dst-CSR (sequential per graph => deterministic)
  bfs_csr_kernel<<<NG, 1, 0, stream>>>(edge_idx, perm, doff, dedge, scratch);

  // 2) h = x @ proj_in_W + b          [8192,256] = [8192,64]x[64,256]
  wmma_gemm_kernel<false><<<dim3(NN / 16, HD / 64), 32, 0, stream>>>(
      x, projWt, proj_in_b, h_buf, NN, HD, 64);

  // 3) e = relu(ea @ W1 + b1) @ W2 + b2
  wmma_gemm_kernel<true><<<dim3(NE / 16, HD / 64), 32, 0, stream>>>(
      ea_pad, edgeW1t, edge_b1, e1, NE, HD, 32);
  wmma_gemm_kernel<false><<<dim3(NE / 16, HD / 64), 32, 0, stream>>>(
      e1, edgeW2t, edge_b2, e_buf, NE, HD, HD);

  // 4) three GINE layers
  for (int l = 0; l < 3; ++l) {
    gine_aggregate_kernel<<<NN, HD, 0, stream>>>(
        h_buf, e_buf, edge_idx, doff, dedge, gine_eps + l, z);
    wmma_gemm_kernel<true><<<dim3(NN / 16, HD / 64), 32, 0, stream>>>(
        z, gineW1t + (size_t)l * HD * HD, gine_b1 + l * HD, t1, NN, HD, HD);
    wmma_gemm_kernel<false><<<dim3(NN / 16, HD / 64), 32, 0, stream>>>(
        t1, gineW2t + (size_t)l * HD * HD, gine_b2 + l * HD, z2, NN, HD, HD);
    bn_stats_kernel<<<HD, 256, 0, stream>>>(z2, stat);
    bn_res_kernel<<<NN, 256, 0, stream>>>(z2, stat, bn_gamma + l * HD, bn_beta + l * HD, h_buf);
  }

  // 5) sequence build + GRU input projection gx = X2 @ Wih^T + bih
  build_X_kernel<<<NN, HD, 0, stream>>>(h_buf, perm, pe, X2);
  wmma_gemm_kernel<false><<<dim3(NN / 16, 768 / 64), 32, 0, stream>>>(
      X2, WihH, gru_bih, gx, NN, 768, HD);

  // 6) GRU recurrence: 128 sequential steps of a tiny WMMA GEMM + gate update
  hipMemsetAsync(hstate, 0, (size_t)NG * HD * sizeof(float), stream);
  for (int t = 0; t < NPG; ++t) {
    wmma_gemm_kernel<false><<<dim3(NG / 16, 768 / 64), 32, 0, stream>>>(
        hstate, WhhH, gru_bhh, gh, NG, 768, HD);
    gru_gate_kernel<<<NG, HD, 0, stream>>>(gx + (size_t)t * NG * 768, gh, hstate, hsum, t);
  }

  // 7) fused = [last | mean_tok | gmp] ; out = fused @ out_W + out_b
  assemble_fused_kernel<<<NG, HD, 0, stream>>>(h_buf, hstate, hsum, fused);
  wmma_gemm_kernel<false><<<dim3(NG / 16, HD / 64), 32, 0, stream>>>(
      fused, outWt, out_b, out, NG, HD, 768);
}
